// GraphAttentionLayer_5720896438795
// MI455X (gfx1250) — matrix operand order repaired and checked by reading, not compiled, not run on hardware
//
#include <hip/hip_runtime.h>
#include <hip/hip_bf16.h>

// ---------- types for CDNA5 WMMA / bf16 conversion ----------
typedef __attribute__((ext_vector_type(16))) __bf16 v16bf;
typedef __attribute__((ext_vector_type(2)))  __bf16 v2bf;
typedef __attribute__((ext_vector_type(8)))  float  v8f;
typedef __attribute__((ext_vector_type(2)))  float  v2f;

union FragAB {
    v16bf v;
    unsigned int u[8];
};

// packed f32x2 -> bf16x2 (RNE) -> u32 ; lowers to native v_cvt_pk_bf16_f32
__device__ __forceinline__ unsigned int pkbf(float x, float y) {
    v2f f = {x, y};
    v2bf b = __builtin_convertvector(f, v2bf);
    return __builtin_bit_cast(unsigned int, b);
}
__device__ __forceinline__ unsigned short f2bf(float f) {
    __bf16 b = (__bf16)f;                 // native cvt, RNE
    return __builtin_bit_cast(unsigned short, b);
}

// =====================================================================
// Kernel 1: per-batch exclusive-rank of adjacency nonzeros (cumsum - 1)
// =====================================================================
__global__ __launch_bounds__(256)
void rank_kernel(const int* __restrict__ adj, int* __restrict__ rank) {
    __shared__ int sc[256];
    const int b = blockIdx.x;
    const int t = threadIdx.x;
    int carry = 0;
    for (int base = 0; base < 96 * 96; base += 256) {
        int v = adj[b * 9216 + base + t];
        __syncthreads();
        sc[t] = v;
        __syncthreads();
        #pragma unroll
        for (int off = 1; off < 256; off <<= 1) {
            int x = (t >= off) ? sc[t - off] : 0;
            __syncthreads();
            sc[t] += x;
            __syncthreads();
        }
        rank[b * 9216 + base + t] = carry + sc[t] - 1;
        carry += sc[255];
    }
}

// =====================================================================
// Kernel 2: C[M,512] = A[M,512] @ W[512,512] via bf16 WMMA.
// Block tile 128x128, 8 waves as 4(M) x 2(N); wave tile 32x64
// (acc[2][4] 16x16 tiles)  ->  8 WMMAs per 12 ds_load_b128 per K-step.
// Double-buffered LDS with register staging of the next K-tile.
//   mode 0: store C                      (g = h @ W_lin)
//   mode 1: fold per-head dot with w3[]  (edot[M,8]); C never stored
// =====================================================================
#define BM 128
#define BN 128
#define BK 32
#define NKT (512 / BK)

__global__ __launch_bounds__(256)
void gemm_bf16(const float* __restrict__ A, const float* __restrict__ W,
               float* __restrict__ C, float* __restrict__ edot,
               const float* __restrict__ w3, int mode) {
    __shared__ __align__(16) unsigned short lA[2][BM * BK]; // [m][k]
    __shared__ __align__(16) unsigned short lB[2][BN * BK]; // [n][k] (transposed)

    const int t     = threadIdx.x;
    const int wave  = t >> 5;            // 0..7
    const int waveM = wave >> 1;         // 0..3 -> rows waveM*32
    const int waveN = wave & 1;          // 0..1 -> cols waveN*64
    const int lane  = t & 31;
    const int ln    = lane & 15;
    const int hi    = lane >> 4;         // lane half (ISA fragment layouts)
    const int gm    = blockIdx.x * BM;
    const int gn    = blockIdx.y * BN;

    v8f acc[2][4] = {};

    float4 ra[4], rb[4];

    // ---- global load of K-tile kt into registers ----
    auto gload = [&](int kt) {
        const int k0 = kt * BK;
        #pragma unroll
        for (int i = 0; i < 4; ++i) {       // A: 128x32 = 1024 float4
            int e4 = i * 256 + t;
            int r  = e4 >> 3;
            int c4 = e4 & 7;
            ra[i] = *(const float4*)&A[(size_t)(gm + r) * 512 + k0 + c4 * 4];
        }
        #pragma unroll
        for (int i = 0; i < 4; ++i) {       // W: 32x128 = 1024 float4
            int e4 = i * 256 + t;
            int kr = e4 >> 5;
            int c4 = e4 & 31;
            rb[i] = *(const float4*)&W[(size_t)(k0 + kr) * 512 + gn + c4 * 4];
        }
    };
    // ---- convert + store registers into LDS buffer buf ----
    auto lstore = [&](int buf) {
        #pragma unroll
        for (int i = 0; i < 4; ++i) {
            int e4 = i * 256 + t;
            int r  = e4 >> 3;
            int c4 = e4 & 7;
            *(unsigned int*)&lA[buf][r * BK + c4 * 4]     = pkbf(ra[i].x, ra[i].y);
            *(unsigned int*)&lA[buf][r * BK + c4 * 4 + 2] = pkbf(ra[i].z, ra[i].w);
        }
        #pragma unroll
        for (int i = 0; i < 4; ++i) {
            int e4 = i * 256 + t;
            int kr = e4 >> 5;
            int c4 = e4 & 31;
            lB[buf][(c4 * 4 + 0) * BK + kr] = f2bf(rb[i].x);
            lB[buf][(c4 * 4 + 1) * BK + kr] = f2bf(rb[i].y);
            lB[buf][(c4 * 4 + 2) * BK + kr] = f2bf(rb[i].z);
            lB[buf][(c4 * 4 + 3) * BK + kr] = f2bf(rb[i].w);
        }
    };

    // prologue: stage tile 0
    gload(0);
    lstore(0);
    __syncthreads();

    for (int kt = 0; kt < NKT; ++kt) {
        const int cur = kt & 1;
        if (kt + 1 < NKT) gload(kt + 1);   // prefetch next tile into registers

        // ---- A fragments: 2 row sub-tiles (ISA 16-bit A 16x32 layout) ----
        FragAB a[2];
        #pragma unroll
        for (int mi = 0; mi < 2; ++mi)
            #pragma unroll
            for (int v = 0; v < 8; ++v) {
                int kb = ((v & 4) << 2) + (hi << 3) + ((v & 3) << 1);
                a[mi].u[v] = *(const unsigned int*)
                    &lA[cur][(waveM * 32 + mi * 16 + ln) * BK + kb];
            }
        // ---- 4 B fragments x 2 A fragments = 8 WMMAs ----
        #pragma unroll
        for (int tt = 0; tt < 4; ++tt) {
            FragAB b;
            #pragma unroll
            for (int v = 0; v < 8; ++v) {
                int kb = ((v & 4) << 2) + (hi << 3) + ((v & 3) << 1);
                b.u[v] = *(const unsigned int*)
                    &lB[cur][(waveN * 64 + tt * 16 + ln) * BK + kb];
            }
            #pragma unroll
            for (int mi = 0; mi < 2; ++mi)
                acc[mi][tt] = __builtin_amdgcn_wmma_f32_16x16x32_bf16(
                    false, a[mi].v, false, b.v, (short)0, acc[mi][tt], false, false);
        }

        if (kt + 1 < NKT) {
            __syncthreads();               // all waves done reading buf[1-cur]
            lstore(1 - cur);
            __syncthreads();               // buf[1-cur] ready for next step
        }
    }

    if (mode == 0) {
        #pragma unroll
        for (int mi = 0; mi < 2; ++mi)
            #pragma unroll
            for (int tt = 0; tt < 4; ++tt)
                #pragma unroll
                for (int v = 0; v < 8; ++v) {
                    int r = gm + waveM * 32 + mi * 16 + hi * 8 + v;
                    int c = gn + waveN * 64 + tt * 16 + ln;
                    C[(size_t)r * 512 + c] = acc[mi][tt][v];
                }
    } else {
        // per-head dot with w3 (wave's 64-col half == exactly one head)
        const int head = blockIdx.y * 2 + waveN;
        float w[4];
        #pragma unroll
        for (int tt = 0; tt < 4; ++tt) w[tt] = w3[tt * 16 + ln];
        #pragma unroll
        for (int mi = 0; mi < 2; ++mi) {
            float p[8];
            #pragma unroll
            for (int v = 0; v < 8; ++v)
                p[v] = acc[mi][0][v] * w[0] + acc[mi][1][v] * w[1] +
                       acc[mi][2][v] * w[2] + acc[mi][3][v] * w[3];
            #pragma unroll
            for (int off = 1; off < 16; off <<= 1)
                #pragma unroll
                for (int v = 0; v < 8; ++v)
                    p[v] += __shfl_xor(p[v], off, 32);
            if (ln == 0) {
                #pragma unroll
                for (int v = 0; v < 8; ++v) {
                    int r = gm + waveM * 32 + mi * 16 + hi * 8 + v;
                    edot[(size_t)r * 8 + head] = p[v];
                }
            }
        }
    }
}

// =====================================================================
// Kernel 3: j-independent dots: ga1/ga2 (g . W_attn halves),
//           os1/os2 (obj . W_score halves). One thread per (b,n,h).
// =====================================================================
__global__ __launch_bounds__(256)
void precompute(const float* __restrict__ g, const float* __restrict__ obj,
                const float* __restrict__ Wa, const float* __restrict__ Ws,
                float* __restrict__ ga1, float* __restrict__ ga2,
                float* __restrict__ os1, float* __restrict__ os2) {
    int idx = blockIdx.x * 256 + threadIdx.x;       // (b*96+n)*8 + h
    if (idx >= 8 * 96 * 8) return;
    int h  = idx & 7;
    int bn = idx >> 3;
    const float* gp = g   + (size_t)bn * 512 + h * 64;
    const float* op = obj + (size_t)bn * 512 + h * 64;
    float s1 = 0.f, s2 = 0.f, s3 = 0.f, s4 = 0.f;
    #pragma unroll 8
    for (int f = 0; f < 64; ++f) {
        float gv = gp[f], ov = op[f];
        s1 += gv * Wa[f];      s2 += gv * Wa[64 + f];
        s3 += ov * Ws[f];      s4 += ov * Ws[64 + f];
    }
    ga1[idx] = s1; ga2[idx] = s2; os1[idx] = s3; os2[idx] = s4;
}

// =====================================================================
// Kernel 4: logits + softmax + out = a @ g.
// One block per (b,i); wave == head h; lane owns j = l, l+32, l+64.
// =====================================================================
__global__ __launch_bounds__(256)
void attention(const float* __restrict__ g, const int* __restrict__ adj,
               const int* __restrict__ rank, const float* __restrict__ edot,
               const float* __restrict__ ga1, const float* __restrict__ ga2,
               const float* __restrict__ os1, const float* __restrict__ os2,
               float* __restrict__ out) {
    __shared__ float aw[8 * 96];
    const int bi = blockIdx.x;           // b*96 + i
    const int b  = bi / 96;
    const int i  = bi % 96;
    const int h  = threadIdx.x >> 5;
    const int l  = threadIdx.x & 31;

    const float g1 = ga1[bi * 8 + h];
    const float o1 = os1[bi * 8 + h];

    float ev[3];
    #pragma unroll
    for (int s = 0; s < 3; ++s) {
        int j    = l + s * 32;
        int aidx = b * 9216 + i * 96 + j;
        int ad   = adj[aidx];
        float e3 = 0.f;
        if (ad != 0) {
            int r = rank[aidx];
            if (r >= 0 && r < 4608)       // rows past E are zero-padded in ref
                e3 = edot[((size_t)b * 4608 + r) * 8 + h];
        }
        float e = g1 + ga2[(b * 96 + j) * 8 + h] + e3;
        e = (e > 0.f) ? e : 0.2f * e;                 // leaky_relu(0.2)
        float w = o1 + os2[(b * 96 + j) * 8 + h];
        w = (w > 0.f) ? w : 0.f;                      // relu
        e *= w;
        ev[s] = (ad != 0) ? e : -1000.0f;
    }
    float mx = fmaxf(ev[0], fmaxf(ev[1], ev[2]));
    #pragma unroll
    for (int off = 16; off >= 1; off >>= 1) mx = fmaxf(mx, __shfl_xor(mx, off, 32));
    float sum = 0.f;
    #pragma unroll
    for (int s = 0; s < 3; ++s) { ev[s] = __expf(ev[s] - mx); sum += ev[s]; }
    #pragma unroll
    for (int off = 16; off >= 1; off >>= 1) sum += __shfl_xor(sum, off, 32);
    const float inv = 1.0f / sum;
    #pragma unroll
    for (int s = 0; s < 3; ++s) aw[h * 96 + l + s * 32] = ev[s] * inv;
    __syncthreads();

    float acc0 = 0.f, acc1 = 0.f;
    for (int j = 0; j < 96; ++j) {
        float aj = aw[h * 96 + j];
        const float* gp = g + (size_t)(b * 96 + j) * 512 + h * 64;
        acc0 += aj * gp[l];
        acc1 += aj * gp[l + 32];
    }
    out[(size_t)bi * 512 + h * 64 + l]      = acc0;
    out[(size_t)bi * 512 + h * 64 + l + 32] = acc1;
}

// =====================================================================
extern "C" void kernel_launch(void* const* d_in, const int* in_sizes, int n_in,
                              void* d_out, int out_size, void* d_ws, size_t ws_size,
                              hipStream_t stream) {
    (void)in_sizes; (void)n_in; (void)out_size; (void)ws_size;
    const float* h_in    = (const float*)d_in[0];
    const int*   adj     = (const int*)  d_in[1];
    const float* edge    = (const float*)d_in[2];
    const float* obj     = (const float*)d_in[3];
    const float* W_lin   = (const float*)d_in[4];
    const float* W_edge  = (const float*)d_in[5];
    const float* W_attn  = (const float*)d_in[6];
    const float* W_score = (const float*)d_in[7];
    float* out = (float*)d_out;

    // workspace layout (~3.2 MB)
    float* g    = (float*)d_ws;            // 768*512
    float* edot = g    + 768 * 512;        // 36864*8
    float* ga1  = edot + 36864 * 8;        // 6144
    float* ga2  = ga1  + 6144;
    float* os1  = ga2  + 6144;
    float* os2  = os1  + 6144;
    int*   rank = (int*)(os2 + 6144);      // 73728

    rank_kernel<<<8, 256, 0, stream>>>(adj, rank);
    // g = h @ W_lin : M=768 rows, N=512 -> grid (6, 4)
    gemm_bf16<<<dim3(6, 4), 256, 0, stream>>>(h_in, W_lin, g, nullptr, nullptr, 0);
    // edot[row,head] = (edge_attr @ W_edge)[row,head,:] . W_attn[128:192]
    gemm_bf16<<<dim3(288, 4), 256, 0, stream>>>(edge, W_edge, nullptr, edot,
                                                W_attn + 128, 1);
    precompute<<<24, 256, 0, stream>>>(g, obj, W_attn, W_score, ga1, ga2, os1, os2);
    attention<<<768, 256, 0, stream>>>(g, adj, rank, edot, ga1, ga2, os1, os2, out);
}
